// CARMN_15315853377661
// MI455X (gfx1250) — matrix operand order repaired and checked by reading, not compiled, not run on hardware
//
#include <hip/hip_runtime.h>
#include <hip/hip_bf16.h>
#include <stdint.h>

// ---------------------------------------------------------------------------
// CARMN forward for MI455X (gfx1250, wave32, WMMA).
// Heavy GEMMs + attention use v_wmma_f32_16x16x32_bf16 (bf16 A/B, f32 acc).
// ---------------------------------------------------------------------------

#define DEV __device__ __forceinline__

typedef __attribute__((ext_vector_type(16))) __bf16 v16bf;
typedef __attribute__((ext_vector_type(8)))  float  v8f;

union Frag16 {
  v16bf v;
  uint4 q[2];
  unsigned short u[16];
};

DEV unsigned short f2bf(float f) {
  unsigned int u = __float_as_uint(f);
  u += 0x7FFFu + ((u >> 16) & 1u);            // round-to-nearest-even
  return (unsigned short)(u >> 16);
}

DEV float leaky(float x) { return x < 0.f ? 0.01f * x : x; }

// Problem constants (hardwired model).
constexpr int Bn  = 64;
constexpr int T   = 291;
constexpr int D   = 300;
constexpr int H   = 10;
constexpr int HK  = 30;        // head dim
constexpr int M   = Bn * T;    // 18624 (multiple of 32)
constexpr int KP  = 320;       // padded K / padded feature width (10 x 32)
constexpr int TP  = 304;       // padded T for attention (19 x 16)
constexpr int MT32 = M / 32;   // 582
constexpr int NT32 = KP / 32;  // 10

enum { MODE_F32 = 0, MODE_BF16PAD = 1, MODE_PACK = 2 };

// ---------------------------------------------------------------------------
// Generic zero fill (bf16 buffers; establishes zero padding invariants).
// ---------------------------------------------------------------------------
__global__ void k_zero_us(unsigned short* p, long long n) {
  long long i = (long long)blockIdx.x * blockDim.x + threadIdx.x;
  long long stride = (long long)gridDim.x * blockDim.x;
  for (; i < n; i += stride) p[i] = 0;
}

// fp32 [D,D] weight -> bf16 [KP,KP] zero-padded
__global__ void k_pad_w(const float* __restrict__ src, unsigned short* __restrict__ dst) {
  int idx = blockIdx.x * blockDim.x + threadIdx.x;
  if (idx >= KP * KP) return;
  int r = idx / KP, c = idx - r * KP;
  dst[idx] = (r < D && c < D) ? f2bf(src[r * D + c]) : (unsigned short)0;
}

// ---------------------------------------------------------------------------
// Image branch: img = leaky(x2 @ W_img^T + b_img)   [B,300]
// ---------------------------------------------------------------------------
__global__ void k_img(const float* __restrict__ x2, const float* __restrict__ Wimg,
                      const float* __restrict__ bimg, float* __restrict__ img) {
  int b = blockIdx.x;
  const float* xr = x2 + (size_t)b * 1000;
  for (int j = threadIdx.x; j < D; j += blockDim.x) {
    float acc = bimg[j];
    const float* wr = Wimg + (size_t)j * 1000;
    for (int i = 0; i < 1000; ++i) acc += xr[i] * wr[i];
    img[b * D + j] = leaky(acc);
  }
}

// LayerNorm over [1,300] -> lnP
__global__ void k_ln_small(const float* __restrict__ img, const float* __restrict__ g,
                           const float* __restrict__ be, float* __restrict__ lnP) {
  int b = blockIdx.x;
  __shared__ float rs[256], rq[256];
  const float* x = img + (size_t)b * D;
  float s = 0.f, q = 0.f;
  for (int i = threadIdx.x; i < D; i += 256) { float v = x[i]; s += v; q += v * v; }
  rs[threadIdx.x] = s; rq[threadIdx.x] = q;
  __syncthreads();
  for (int st = 128; st > 0; st >>= 1) {
    if (threadIdx.x < st) { rs[threadIdx.x] += rs[threadIdx.x + st]; rq[threadIdx.x] += rq[threadIdx.x + st]; }
    __syncthreads();
  }
  float mu = rs[0] / D;
  float rstd = rsqrtf(rq[0] / D - mu * mu + 1e-5f);
  for (int i = threadIdx.x; i < D; i += 256)
    lnP[b * D + i] = (x[i] - mu) * rstd * g[i] + be[i];
}

// LayerNorm over [291,300] -> lnT (fp32) and bf16 padded copy [M,KP]
__global__ void k_ln_big(const float* __restrict__ x1, const float* __restrict__ g,
                         const float* __restrict__ be, float* __restrict__ lnT,
                         unsigned short* __restrict__ xbf) {
  int b = blockIdx.x;
  const int n = T * D; // 87300
  const float* x = x1 + (size_t)b * n;
  __shared__ float rs[256], rq[256];
  float s = 0.f, q = 0.f;
  for (int i = threadIdx.x; i < n; i += 256) { float v = x[i]; s += v; q += v * v; }
  rs[threadIdx.x] = s; rq[threadIdx.x] = q;
  __syncthreads();
  for (int st = 128; st > 0; st >>= 1) {
    if (threadIdx.x < st) { rs[threadIdx.x] += rs[threadIdx.x + st]; rq[threadIdx.x] += rq[threadIdx.x + st]; }
    __syncthreads();
  }
  float mu = rs[0] / n;
  float rstd = rsqrtf(rq[0] / n - mu * mu + 1e-5f);
  for (int i = threadIdx.x; i < n; i += 256) {
    int t = i / D, d = i - t * D;
    float v = (x[i] - mu) * rstd * g[i] + be[i];
    lnT[(size_t)b * n + i] = v;
    xbf[((size_t)(b * T + t)) * KP + d] = f2bf(v);
  }
  // zero k-padding columns 300..319
  for (int i = threadIdx.x; i < T * (KP - D); i += 256) {
    int t = i / (KP - D), d = D + (i - t * (KP - D));
    xbf[((size_t)(b * T + t)) * KP + d] = 0;
  }
}

// ---------------------------------------------------------------------------
// Cross-attention branches, algebraically reduced (single query / singleton
// softmax). One block per batch. Also produces vpt = Wv_pt @ lnP.
// ---------------------------------------------------------------------------
__global__ void k_tp(const float* __restrict__ lnP, const float* __restrict__ lnT,
                     const float* __restrict__ Wq, const float* __restrict__ Wk,
                     const float* __restrict__ Wv, const float* __restrict__ Wvpt,
                     float* __restrict__ rtp, float* __restrict__ vpt) {
  int b = blockIdx.x;
  __shared__ float sP[D], qp[D], kq[D], ctx[D], sc[T], red[256];
  int tid = threadIdx.x;
  for (int j = tid; j < D; j += 256) sP[j] = lnP[b * D + j];
  __syncthreads();
  for (int j = tid; j < D; j += 256) {          // qp = Wq lnP
    float a = 0.f;
    const float* wr = Wq + (size_t)j * D;
    for (int i = 0; i < D; ++i) a += wr[i] * sP[i];
    qp[j] = a;
  }
  __syncthreads();
  for (int j = tid; j < D; j += 256) {          // kq = Wk^T qp
    float a = 0.f;
    for (int i = 0; i < D; ++i) a += Wk[(size_t)i * D + j] * qp[i];
    kq[j] = a;
  }
  __syncthreads();
  const float* xT = lnT + (size_t)b * T * D;
  for (int t = tid; t < T; t += 256) {          // scores
    float a = 0.f;
    const float* xr = xT + (size_t)t * D;
    for (int i = 0; i < D; ++i) a += xr[i] * kq[i];
    sc[t] = a;
  }
  __syncthreads();
  float mx = -3.0e38f;
  for (int t = tid; t < T; t += 256) mx = fmaxf(mx, sc[t]);
  red[tid] = mx; __syncthreads();
  for (int st = 128; st > 0; st >>= 1) { if (tid < st) red[tid] = fmaxf(red[tid], red[tid + st]); __syncthreads(); }
  mx = red[0]; __syncthreads();
  float sm = 0.f;
  for (int t = tid; t < T; t += 256) { float p = __expf(sc[t] - mx); sc[t] = p; sm += p; }
  red[tid] = sm; __syncthreads();
  for (int st = 128; st > 0; st >>= 1) { if (tid < st) red[tid] += red[tid + st]; __syncthreads(); }
  float inv = 1.0f / red[0];
  __syncthreads();
  for (int t = tid; t < T; t += 256) sc[t] *= inv;
  __syncthreads();
  for (int d = tid; d < D; d += 256) {          // ctx = sum_t a_t lnT_t
    float a = 0.f;
    for (int t = 0; t < T; ++t) a += sc[t] * xT[(size_t)t * D + d];
    ctx[d] = a;
  }
  __syncthreads();
  for (int j = tid; j < D; j += 256) {          // R_T_P and vpt
    float a = 0.f, av = 0.f;
    const float* wr  = Wv   + (size_t)j * D;
    const float* wr2 = Wvpt + (size_t)j * D;
    for (int i = 0; i < D; ++i) { a += wr[i] * ctx[i]; av += wr2[i] * sP[i]; }
    rtp[b * D + j] = a + sP[j];
    vpt[b * D + j] = av;
  }
}

// ---------------------------------------------------------------------------
// WMMA GEMM: C[M,N] = A[M,Kp] * Bw[N,Kp]^T   (bf16 in, f32 acc)
// Wave computes a 32x32 tile. Epilogue modes:
//   MODE_F32     : fp32 [M,300]
//   MODE_BF16PAD : bf16 [M,320] (zero padded) -> next block's input
//   MODE_PACK    : bf16 head-packed [B,H,TP,32] with scale -> attention input
// ---------------------------------------------------------------------------
DEV void gemm_emit(int mode, float scale, void* outp, int row, int col, float val) {
  val *= scale;
  if (mode == MODE_F32) {
    if (col < D) ((float*)outp)[(size_t)row * D + col] = val;
  } else if (mode == MODE_BF16PAD) {
    ((unsigned short*)outp)[(size_t)row * KP + col] = (col < D) ? f2bf(val) : (unsigned short)0;
  } else {
    if (col < D) {
      int b = row / T, t = row - b * T;
      int h = col / HK, k = col - h * HK;
      ((unsigned short*)outp)[(((size_t)(b * H + h) * TP) + t) * 32 + k] = f2bf(val);
    }
  }
}

__global__ void k_gemm(const unsigned short* __restrict__ A,
                       const unsigned short* __restrict__ Bw,
                       void* __restrict__ outp, int mode, float scale) {
  int w = blockIdx.x * (blockDim.x >> 5) + (threadIdx.x >> 5);
  if (w >= MT32 * NT32) return;
  int tm = w % MT32, tn = w / MT32;
  int lane = threadIdx.x & 31;
  int hlf  = lane >> 4;
  int l16  = lane & 15;
  int kh8  = hlf * 8;
  int kOff = hlf * 16;

  const unsigned short* a0p = A + (size_t)(tm * 32 + l16) * KP;
  const unsigned short* a1p = a0p + (size_t)16 * KP;
  const unsigned short* b0p = Bw + (size_t)(tn * 32 + l16) * KP + kOff;
  const unsigned short* b1p = b0p + (size_t)16 * KP;

  v8f c00 = {}, c01 = {}, c10 = {}, c11 = {};
#pragma unroll 2
  for (int k0 = 0; k0 < KP; k0 += 32) {
    Frag16 a0, a1, b0, b1;
    a0.q[0] = *(const uint4*)(a0p + k0 + kOff);
    a0.q[1] = *(const uint4*)(a0p + k0 + kOff + 8);
    a1.q[0] = *(const uint4*)(a1p + k0 + kOff);
    a1.q[1] = *(const uint4*)(a1p + k0 + kOff + 8);
    b0.q[0] = *(const uint4*)(b0p + k0);
    b0.q[1] = *(const uint4*)(b0p + k0 + 8);
    b1.q[0] = *(const uint4*)(b1p + k0);
    b1.q[1] = *(const uint4*)(b1p + k0 + 8);
    c00 = __builtin_amdgcn_wmma_f32_16x16x32_bf16(false, a0.v, false, b0.v, (short)0, c00, false, false);
    c01 = __builtin_amdgcn_wmma_f32_16x16x32_bf16(false, a0.v, false, b1.v, (short)0, c01, false, false);
    c10 = __builtin_amdgcn_wmma_f32_16x16x32_bf16(false, a1.v, false, b0.v, (short)0, c10, false, false);
    c11 = __builtin_amdgcn_wmma_f32_16x16x32_bf16(false, a1.v, false, b1.v, (short)0, c11, false, false);
  }
  int rowBase = tm * 32 + hlf * 8;
  int colBase = tn * 32 + l16;
#pragma unroll
  for (int r = 0; r < 8; ++r) {
    gemm_emit(mode, scale, outp, rowBase + r,      colBase,      c00[r]);
    gemm_emit(mode, scale, outp, rowBase + r,      colBase + 16, c01[r]);
    gemm_emit(mode, scale, outp, rowBase + 16 + r, colBase,      c10[r]);
    gemm_emit(mode, scale, outp, rowBase + 16 + r, colBase + 16, c11[r]);
  }
}

// ---------------------------------------------------------------------------
// Fused per-(b,h,i-tile) attention: scores (WMMA) -> masked softmax
// (half-wave shfl reductions) -> LDS transpose -> att @ V (WMMA) ->
// head-packed bf16 output AO[M,KP].
// ---------------------------------------------------------------------------
__global__ void k_attn(const unsigned short* __restrict__ qh,
                       const unsigned short* __restrict__ khd,
                       const unsigned short* __restrict__ vh,
                       unsigned short* __restrict__ AO) {
  __shared__ unsigned short att_s[4][16][KP];   // 40 KB
  int wv = threadIdx.x >> 5;
  int w = blockIdx.x * 4 + wv;
  const int NWAVES = Bn * H * (TP / 16);        // 12160
  if (w >= NWAVES) return;
  int it = w % (TP / 16);
  int bh = w / (TP / 16);                       // b*H + h
  int lane = threadIdx.x & 31;
  int hlf  = lane >> 4;
  int l16  = lane & 15;
  int kh8  = hlf * 8;
  int kOff = hlf * 16;

  const unsigned short* qbase = qh  + (size_t)bh * TP * 32;
  const unsigned short* kbase = khd + (size_t)bh * TP * 32;
  const unsigned short* vbase = vh  + (size_t)bh * TP * 32;

  // A fragment of q for this i-tile (head dim 30 padded to K=32, one shot)
  Frag16 aq;
  {
    const unsigned short* p = qbase + (size_t)(it * 16 + l16) * 32;
    aq.q[0] = *(const uint4*)(p + kOff);
    aq.q[1] = *(const uint4*)(p + kOff + 8);
  }

  // scores: 19 j-tiles of 16
  v8f s[19];
#pragma unroll
  for (int jt = 0; jt < 19; ++jt) {
    Frag16 bk;
    const unsigned short* p = kbase + (size_t)(jt * 16 + l16) * 32 + kOff;
    bk.q[0] = *(const uint4*)p;
    bk.q[1] = *(const uint4*)(p + 8);
    v8f z = {};
    s[jt] = __builtin_amdgcn_wmma_f32_16x16x32_bf16(false, aq.v, false, bk.v, (short)0, z, false, false);
  }

  // masked softmax per row; C layout: row = hlf*8 + r, col = l16 within tile
#pragma unroll
  for (int r = 0; r < 8; ++r) {
    int rowL = hlf * 8 + r;
    float vbuf[19];
    float mx = -3.0e38f;
#pragma unroll
    for (int jt = 0; jt < 19; ++jt) {
      int j = jt * 16 + l16;
      float x = (j < T) ? s[jt][r] : -1.0e30f;
      vbuf[jt] = x;
      mx = fmaxf(mx, x);
    }
    mx = fmaxf(mx, __shfl_xor(mx, 1, 32));
    mx = fmaxf(mx, __shfl_xor(mx, 2, 32));
    mx = fmaxf(mx, __shfl_xor(mx, 4, 32));
    mx = fmaxf(mx, __shfl_xor(mx, 8, 32));
    float sum = 0.f;
#pragma unroll
    for (int jt = 0; jt < 19; ++jt) { float p = __expf(vbuf[jt] - mx); vbuf[jt] = p; sum += p; }
    sum += __shfl_xor(sum, 1, 32);
    sum += __shfl_xor(sum, 2, 32);
    sum += __shfl_xor(sum, 4, 32);
    sum += __shfl_xor(sum, 8, 32);
    float inv = 1.0f / sum;
#pragma unroll
    for (int jt = 0; jt < 19; ++jt)
      att_s[wv][rowL][jt * 16 + l16] = f2bf(vbuf[jt] * inv);
  }
  // zero LDS padding columns 304..319
  for (int idx = lane; idx < 16 * 16; idx += 32)
    att_s[wv][idx >> 4][304 + (idx & 15)] = 0;

  // O = att @ V  (att is 16x320 in LDS, V is 304x32 head-packed)
#pragma unroll
  for (int nt = 0; nt < 2; ++nt) {
    v8f acc = {};
    for (int j0 = 0; j0 < KP; j0 += 32) {
      Frag16 aa, bv;
      const unsigned short* p = &att_s[wv][l16][j0];
      aa.q[0] = *(const uint4*)(p + kOff);
      aa.q[1] = *(const uint4*)(p + kOff + 8);
#pragma unroll
      for (int e = 0; e < 16; ++e) {
        int t = j0 + kOff + e;
        bv.u[e] = (t < TP) ? vbase[(size_t)t * 32 + nt * 16 + l16] : (unsigned short)0;
      }
      acc = __builtin_amdgcn_wmma_f32_16x16x32_bf16(false, aa.v, false, bv.v, (short)0, acc, false, false);
    }
#pragma unroll
    for (int r = 0; r < 8; ++r) {
      int i = it * 16 + hlf * 8 + r;
      int kk = nt * 16 + l16;
      if (i < T && kk < HK) {
        int b = bh / H, h = bh - b * H;
        AO[((size_t)(b * T + i)) * KP + h * HK + kk] = f2bf(acc[r]);
      }
    }
  }
}

// ---------------------------------------------------------------------------
// Conv head: block per (b,o); channel 0 = lnT + vpt (R_P_T on the fly),
// channel 1 = a3. leaky + global max over L.
// ---------------------------------------------------------------------------
__global__ void k_conv(const float* __restrict__ lnT, const float* __restrict__ vpt,
                       const float* __restrict__ a3, const float* __restrict__ Wc,
                       const float* __restrict__ bc, float* __restrict__ cout,
                       int khh, int outOff) {
  int bo = blockIdx.x;
  int b = bo / 25, o = bo - b * 25;
  int L = T - khh + 1;
  __shared__ float red[128];
  float mx = -3.0e38f;
  const float* xT = lnT + (size_t)b * T * D;
  const float* aT = a3  + (size_t)b * T * D;
  const float* vp = vpt + (size_t)b * D;
  for (int l = threadIdx.x; l < L; l += blockDim.x) {
    float acc = bc[o];
    for (int i = 0; i < khh; ++i) {
      const float* w0 = Wc + ((size_t)(o * 2 + 0) * khh + i) * D;
      const float* w1 = Wc + ((size_t)(o * 2 + 1) * khh + i) * D;
      const float* r0 = xT + (size_t)(l + i) * D;
      const float* r1 = aT + (size_t)(l + i) * D;
      for (int d = 0; d < D; ++d) acc += (r0[d] + vp[d]) * w0[d] + r1[d] * w1[d];
    }
    mx = fmaxf(mx, leaky(acc));
  }
  red[threadIdx.x] = mx;
  __syncthreads();
  for (int st = 64; st > 0; st >>= 1) {
    if (threadIdx.x < st) red[threadIdx.x] = fmaxf(red[threadIdx.x], red[threadIdx.x + st]);
    __syncthreads();
  }
  if (threadIdx.x == 0) cout[b * 100 + outOff + o] = red[0];
}

// total = leaky(conv_out @ Wlin^T + blin)  [B,300]
__global__ void k_lin(const float* __restrict__ cin, const float* __restrict__ Wlin,
                      const float* __restrict__ blin, float* __restrict__ lout) {
  int idx = blockIdx.x * blockDim.x + threadIdx.x;
  if (idx >= Bn * D) return;
  int b = idx / D, j = idx - b * D;
  float a = blin[j];
  const float* wr = Wlin + (size_t)j * 100;
  const float* cr = cin + (size_t)b * 100;
  for (int i = 0; i < 100; ++i) a += cr[i] * wr[i];
  lout[idx] = leaky(a);
}

// Y = [rtp | total] @ Wlast^T + blast  [B,2]
__global__ void k_last(const float* __restrict__ rtp, const float* __restrict__ lout,
                       const float* __restrict__ Wl, const float* __restrict__ bl,
                       float* __restrict__ y) {
  int idx = threadIdx.x;
  if (idx >= Bn * 2) return;
  int b = idx >> 1, c = idx & 1;
  float a = bl[c];
  const float* wr = Wl + (size_t)c * 600;
  for (int j = 0; j < D; ++j) a += rtp[b * D + j] * wr[j];
  for (int j = 0; j < D; ++j) a += lout[b * D + j] * wr[300 + j];
  y[idx] = a;
}

// ---------------------------------------------------------------------------
extern "C" void kernel_launch(void* const* d_in, const int* in_sizes, int n_in,
                              void* d_out, int out_size, void* d_ws, size_t ws_size,
                              hipStream_t stream) {
  const float* x1    = (const float*)d_in[0];
  const float* x2    = (const float*)d_in[1];
  const float* W_img = (const float*)d_in[2];
  const float* b_img = (const float*)d_in[3];
  const float* g1    = (const float*)d_in[4];
  const float* be1   = (const float*)d_in[5];
  const float* g2    = (const float*)d_in[6];
  const float* be2   = (const float*)d_in[7];
  const float* Wq_tp = (const float*)d_in[8];
  const float* Wk_tp = (const float*)d_in[9];
  const float* Wv_tp = (const float*)d_in[10];
  // d_in[11], d_in[12] (Wq_pt, Wk_pt) unused: softmax over singleton axis == 1
  const float* Wv_pt = (const float*)d_in[13];
  const float* Wc[4] = {(const float*)d_in[26], (const float*)d_in[27],
                        (const float*)d_in[28], (const float*)d_in[29]};
  const float* bc[4] = {(const float*)d_in[30], (const float*)d_in[31],
                        (const float*)d_in[32], (const float*)d_in[33]};
  const float* Wlin  = (const float*)d_in[34];
  const float* blin  = (const float*)d_in[35];
  const float* Wlast = (const float*)d_in[36];
  const float* blast = (const float*)d_in[37];

  char* ws = (char*)d_ws;
  size_t off = 0;
  auto alloc = [&](size_t bytes) -> void* {
    size_t cur = (off + 255) & ~(size_t)255;
    off = cur + bytes;
    return (void*)(ws + cur);
  };

  float* f_lnT  = (float*)alloc((size_t)M * D * 4);
  float* f_a3   = (float*)alloc((size_t)M * D * 4);
  unsigned short* bf_x0 = (unsigned short*)alloc((size_t)M * KP * 2);
  unsigned short* bf_x1 = (unsigned short*)alloc((size_t)M * KP * 2);
  unsigned short* bf_ao = (unsigned short*)alloc((size_t)M * KP * 2);
  unsigned short* bf_q  = (unsigned short*)alloc((size_t)Bn * H * TP * 32 * 2);
  unsigned short* bf_k  = (unsigned short*)alloc((size_t)Bn * H * TP * 32 * 2);
  unsigned short* bf_v  = (unsigned short*)alloc((size_t)Bn * H * TP * 32 * 2);
  float* f_img  = (float*)alloc((size_t)Bn * D * 4);
  float* f_lnP  = (float*)alloc((size_t)Bn * D * 4);
  float* f_rtp  = (float*)alloc((size_t)Bn * D * 4);
  float* f_vpt  = (float*)alloc((size_t)Bn * D * 4);
  float* f_conv = (float*)alloc((size_t)Bn * 100 * 4);
  float* f_lin  = (float*)alloc((size_t)Bn * D * 4);
  unsigned short* bf_w[12];
  for (int i = 0; i < 12; ++i) bf_w[i] = (unsigned short*)alloc((size_t)KP * KP * 2);
  // total ~121 MB

  // --- weight conversion (fp32 -> bf16 padded), once per call ---
  for (int i = 0; i < 12; ++i)
    k_pad_w<<<(KP * KP + 255) / 256, 256, 0, stream>>>((const float*)d_in[14 + i], bf_w[i]);

  // establish zero-padding invariants in packed buffers
  long long nqkv = (long long)Bn * H * TP * 32;
  k_zero_us<<<4096, 256, 0, stream>>>(bf_q, nqkv);
  k_zero_us<<<4096, 256, 0, stream>>>(bf_k, nqkv);
  k_zero_us<<<4096, 256, 0, stream>>>(bf_v, nqkv);
  k_zero_us<<<4096, 256, 0, stream>>>(bf_ao, (long long)M * KP);

  // --- front end ---
  k_img<<<Bn, 256, 0, stream>>>(x2, W_img, b_img, f_img);
  k_ln_small<<<Bn, 256, 0, stream>>>(f_img, g2, be2, f_lnP);
  k_ln_big<<<Bn, 256, 0, stream>>>(x1, g1, be1, f_lnT, bf_x0);
  k_tp<<<Bn, 256, 0, stream>>>(f_lnP, f_lnT, Wq_tp, Wk_tp, Wv_tp, Wv_pt, f_rtp, f_vpt);

  // --- 3 stacked MHA blocks (WMMA) ---
  const int gemm_blocks = (MT32 * NT32 + 7) / 8;
  const float inv_s = 0.5623413251903491f;  // H^-0.25
  unsigned short* xin = bf_x0;
  unsigned short* xout = bf_x1;
  for (int blk = 0; blk < 3; ++blk) {
    const unsigned short* Wq = bf_w[blk * 4 + 0];
    const unsigned short* Wk = bf_w[blk * 4 + 1];
    const unsigned short* Wv = bf_w[blk * 4 + 2];
    const unsigned short* Wm = bf_w[blk * 4 + 3];
    k_gemm<<<gemm_blocks, 256, 0, stream>>>(xin, Wq, bf_q, MODE_PACK, inv_s);
    k_gemm<<<gemm_blocks, 256, 0, stream>>>(xin, Wk, bf_k, MODE_PACK, inv_s);
    k_gemm<<<gemm_blocks, 256, 0, stream>>>(xin, Wv, bf_v, MODE_PACK, 1.0f);
    k_attn<<<(Bn * H * (TP / 16)) / 4, 128, 0, stream>>>(bf_q, bf_k, bf_v, bf_ao);
    if (blk < 2) {
      k_gemm<<<gemm_blocks, 256, 0, stream>>>(bf_ao, Wm, xout, MODE_BF16PAD, 1.0f);
      unsigned short* tmp = xin; xin = xout; xout = tmp;
    } else {
      k_gemm<<<gemm_blocks, 256, 0, stream>>>(bf_ao, Wm, f_a3, MODE_F32, 1.0f);
    }
  }

  // --- conv heads + tail ---
  for (int hh = 0; hh < 4; ++hh)
    k_conv<<<Bn * 25, 128, 0, stream>>>(f_lnT, f_vpt, f_a3, Wc[hh], bc[hh], f_conv,
                                        hh + 1, hh * 25);
  k_lin<<<(Bn * D + 255) / 256, 256, 0, stream>>>(f_conv, Wlin, blin, f_lin);
  k_last<<<1, 128, 0, stream>>>(f_rtp, f_lin, Wlast, blast, (float*)d_out);
}
